// PAM_Module_5832565588565
// MI455X (gfx1250) — matrix operand order repaired and checked by reading, not compiled, not run on hardware
//
#include <hip/hip_runtime.h>
#include <hip/hip_bf16.h>

#define BB  4
#define CC  256
#define NN  4096
#define CKK 32

typedef __attribute__((ext_vector_type(16))) __bf16 v16bf;
typedef __attribute__((ext_vector_type(8)))  __bf16 v8bf;
typedef __attribute__((ext_vector_type(8)))  float  v8f;
typedef __attribute__((ext_vector_type(4)))  unsigned int v4u;
typedef __attribute__((ext_vector_type(8)))  int v8i;
typedef __attribute__((ext_vector_type(4)))  int v4i;

#ifndef __has_builtin
#define __has_builtin(x) 0
#endif
#if __has_builtin(__builtin_amdgcn_tensor_load_to_lds)
#define HAS_TDM 1
#if __has_include(<hip/amd_detail/amd_gfx1250_TDM.h>)
#define TDM_6ARG 1
#endif
#endif

// Load 16 bf16 as two 16B chunks at p and p+off2 (off2=8 -> contiguous 32B,
// off2=16 -> A-operand chunk pattern {0..7}{16..23}).
static __device__ inline v16bf ld_pair(const __bf16* p, int off2) {
  v8bf a = *(const v8bf*)p;
  v8bf b = *(const v8bf*)(p + off2);
  v16bf r;
#pragma unroll
  for (int i = 0; i < 8; ++i) { r[i] = a[i]; r[i + 8] = b[i]; }
  return r;
}

// out[o][n] (+bias) = sum_c W[o][c] * x[b][c][n], bf16 output.
// trans=1: store [B][N][O] (row n contiguous in o)  -- used for Q and K^T
// trans=0: store [B][O][N]                          -- used for V
__global__ __launch_bounds__(256)
void proj_kernel(const float* __restrict__ x, const float* __restrict__ W,
                 const float* __restrict__ bias, __bf16* __restrict__ out,
                 int O, int trans)
{
  const int b    = blockIdx.y;
  const int wave = threadIdx.x >> 5;
  const int lane = threadIdx.x & 31;
  const int hi   = (lane >> 4) & 1;
  const int l    = lane & 15;
  const int tile = blockIdx.x * 8 + wave;
  const int oT   = O >> 4;
  const int o0   = (tile % oT) << 4;
  const int n0   = (tile / oT) << 4;
  const int n    = n0 + l;

  const float* xb   = x + (size_t)b * CC * NN;
  const float* wrow = W + (size_t)(o0 + l) * CC + (hi ? 8 : 0);
  const int    cB0  = hi ? 8 : 0;

  v8f acc = {};
  for (int c0 = 0; c0 < CC; c0 += 32) {
    v16bf a, bm;
    const float* wA = wrow + c0;
#pragma unroll
    for (int i = 0; i < 8; ++i) {
      a[i]     = (__bf16)wA[i];
      a[i + 8] = (__bf16)wA[16 + i];
    }
    const float* xB = xb + (size_t)(c0 + cB0) * NN + n;
#pragma unroll
    for (int i = 0; i < 16; ++i) bm[i] = (__bf16)xB[(size_t)(i + (i & 8)) * NN];
    acc = __builtin_amdgcn_wmma_f32_16x16x32_bf16(false, a, false, bm,
                                                  (short)0, acc, false, false);
  }

  const int rbase = hi ? 8 : 0;
  if (trans) {
    __bf16* dst = out + ((size_t)b * NN + n) * O + o0 + rbase;
    v8bf pk;
#pragma unroll
    for (int e = 0; e < 8; ++e) pk[e] = (__bf16)(acc[e] + bias[o0 + rbase + e]);
    *(v8bf*)dst = pk;
  } else {
#pragma unroll
    for (int e = 0; e < 8; ++e) {
      const int o = o0 + rbase + e;
      out[((size_t)b * O + o) * NN + n] = (__bf16)(acc[e] + bias[o]);
    }
  }
}

// Fused flash attention + gamma*out + x.
// 128-thread block (4 waves) owns 32 query rows (2 n-tiles) of batch blockIdx.y.
// Per 128-key strip: TDM stages K to LDS; wave w computes S for key chunk w for
// both n-tiles (4 WMMA), softmax stats merged via LDS, both P tiles staged in
// LDS, then each wave runs 32 PV WMMAs on its 64-channel slice — every V
// B-operand load is shared by the two n-tiles (halves L2 traffic).
__global__ __launch_bounds__(128)
void attn_kernel(const float* __restrict__ x, const __bf16* __restrict__ Q,
                 const __bf16* __restrict__ Kt, const __bf16* __restrict__ V,
                 const float* __restrict__ gamma, float* __restrict__ out)
{
  __shared__ __align__(16) __bf16 Klds[128][40];  // 80B rows: conflict-free b128 reads
  __shared__ float Pf[2][16][132];                // P tiles (2 x 16 x 128) + pad
  __shared__ float mxs[2][4][16];                 // per-tile per-chunk row maxes
  __shared__ float sms[2][4][16];                 // per-tile per-chunk row sums

  const int wave = threadIdx.x >> 5;
  const int lane = threadIdx.x & 31;
  const int hi   = (lane >> 4) & 1;
  const int l    = lane & 15;
  const int n0   = blockIdx.x << 5;   // 32 query rows per block
  const int b    = blockIdx.y;
  const int c0   = wave << 6;         // 64-channel slice per wave
  const float g  = gamma[0];

  const __bf16* qb = Q  + (size_t)b * NN * CKK;
  const __bf16* kb = Kt + (size_t)b * NN * CKK;
  const __bf16* vb = V  + (size_t)b * CC * NN;

  // Q A-operands for the two 16-row tiles
  v16bf qa[2];
  qa[0] = ld_pair(qb + (size_t)(n0 + l) * CKK + (hi ? 16 : 0), 8);
  qa[1] = ld_pair(qb + (size_t)(n0 + 16 + l) * CKK + (hi ? 16 : 0), 8);

  v8f oacc[2][4] = {};
  float rmax[2][8], rsum[2][8];
#pragma unroll
  for (int t = 0; t < 2; ++t)
#pragma unroll
    for (int e = 0; e < 8; ++e) { rmax[t][e] = -1e30f; rsum[t][e] = 0.f; }

  for (int m0 = 0; m0 < NN; m0 += 128) {
    // ---- Stage K strip [m0 .. m0+127] x 32 into LDS ----
#if HAS_TDM
    if (threadIdx.x < 32) {
      const unsigned long long ga =
          (unsigned long long)(uintptr_t)(kb + (size_t)m0 * CKK);
      const unsigned int ldsa = (unsigned int)(uintptr_t)&Klds[0][0];
      v4u g0;
      g0[0] = 1u;                                   // count=1, user D#
      g0[1] = ldsa;                                 // lds_addr
      g0[2] = (unsigned int)ga;                     // global_addr[31:0]
      g0[3] = ((unsigned int)(ga >> 32) & 0x01ffffffu) | (2u << 30);  // addr[56:32], type=2
      v8i g1;
      // data_size=2B, pad_enable, pad_interval=16 dwords, pad_amount=4 dwords
      g1[0] = (int)((1u << 16) | (1u << 20) | (3u << 22) | (3u << 25));
      g1[1] = (int)(32u << 16);                     // tensor_dim0 = 32 (lo16 @ [63:48])
      g1[2] = (int)(((unsigned)(NN - m0) & 0xffffu) << 16);       // dim0 hi | tensor_dim1 lo
      g1[3] = (int)((((unsigned)(NN - m0)) >> 16) | (32u << 16)); // dim1 hi | tile_dim0=32
      g1[4] = (int)128;                             // tile_dim1=128, tile_dim2=0
      g1[5] = (int)32;                              // tensor_dim0_stride = 32
      g1[6] = 0;
      g1[7] = 0;
      v4i z4 = {0, 0, 0, 0};
#ifdef TDM_6ARG
      v8i z8 = {0, 0, 0, 0, 0, 0, 0, 0};
      __builtin_amdgcn_tensor_load_to_lds(g0, g1, z4, z4, z8, 0);
#else
      __builtin_amdgcn_tensor_load_to_lds(g0, g1, z4, z4, 0);
#endif
#if __has_builtin(__builtin_amdgcn_s_wait_tensorcnt)
      __builtin_amdgcn_s_wait_tensorcnt(0);
#endif
    }
#else
    {
      const int t = threadIdx.x;  // one 64B row per thread
      const __bf16* src = kb + (size_t)(m0 + t) * CKK;
      __bf16* dst = &Klds[t][0];
#pragma unroll
      for (int i = 0; i < 4; ++i)
        *(v8bf*)(dst + i * 8) = *(const v8bf*)(src + i * 8);
    }
#endif
    __syncthreads();  // K strip visible; prev-iter P reads complete

    // ---- S for this wave's 32-key chunk, both n-tiles ----
    const v16bf k0 = ld_pair(&Klds[wave * 32 + l][hi ? 16 : 0], 8);
    const v16bf k1 = ld_pair(&Klds[wave * 32 + 16 + l][hi ? 16 : 0], 8);
    v8f s[2][2] = {};
#pragma unroll
    for (int t = 0; t < 2; ++t) {
      s[t][0] = __builtin_amdgcn_wmma_f32_16x16x32_bf16(false, qa[t], false, k0,
                                                        (short)0, s[t][0], false, false);
      s[t][1] = __builtin_amdgcn_wmma_f32_16x16x32_bf16(false, qa[t], false, k1,
                                                        (short)0, s[t][1], false, false);
    }

#pragma unroll
    for (int t = 0; t < 2; ++t)
#pragma unroll
      for (int e = 0; e < 8; ++e) {
        float m = fmaxf(s[t][0][e], s[t][1][e]);
#pragma unroll
        for (int msk = 1; msk < 16; msk <<= 1) m = fmaxf(m, __shfl_xor(m, msk, 32));
        if (l == 0) mxs[t][wave][e + hi * 8] = m;
      }
    __syncthreads();  // all chunk maxes posted

    // ---- Merge maxes, write P tiles + chunk sums ----
    float sc[2][8];
#pragma unroll
    for (int t = 0; t < 2; ++t)
#pragma unroll
      for (int e = 0; e < 8; ++e) {
        const int row = e + hi * 8;
        float gm = fmaxf(fmaxf(mxs[t][0][row], mxs[t][1][row]),
                         fmaxf(mxs[t][2][row], mxs[t][3][row]));
        const float nm = fmaxf(rmax[t][e], gm);
        sc[t][e]   = __expf(rmax[t][e] - nm);
        rmax[t][e] = nm;

        const float p0 = __expf(s[t][0][e] - nm);
        const float p1 = __expf(s[t][1][e] - nm);
        Pf[t][row][wave * 32 + l]      = p0;
        Pf[t][row][wave * 32 + 16 + l] = p1;
        float rs = p0 + p1;
#pragma unroll
        for (int msk = 1; msk < 16; msk <<= 1) rs += __shfl_xor(rs, msk, 32);
        if (l == 0) sms[t][wave][row] = rs;
      }
    __syncthreads();  // full P tiles + sums posted

    // ---- Update running sums, rescale O ----
#pragma unroll
    for (int t = 0; t < 2; ++t) {
#pragma unroll
      for (int e = 0; e < 8; ++e) {
        const int row = e + hi * 8;
        rsum[t][e] = rsum[t][e] * sc[t][e] +
                     (sms[t][0][row] + sms[t][1][row] + sms[t][2][row] + sms[t][3][row]);
      }
#pragma unroll
      for (int ct = 0; ct < 4; ++ct)
#pragma unroll
        for (int e = 0; e < 8; ++e) oacc[t][ct][e] *= sc[t][e];
    }

    // ---- PV over the 128-wide strip: each V load feeds both n-tiles ----
#pragma unroll
    for (int sub = 0; sub < 4; ++sub) {
      v16bf pa0, pa1;
      const int cb = sub * 32 + (hi ? 8 : 0);
#pragma unroll
      for (int i = 0; i < 8; ++i) {
        pa0[i]     = (__bf16)Pf[0][l][cb + i];
        pa0[i + 8] = (__bf16)Pf[0][l][cb + 16 + i];
        pa1[i]     = (__bf16)Pf[1][l][cb + i];
        pa1[i + 8] = (__bf16)Pf[1][l][cb + 16 + i];
      }
#pragma unroll
      for (int ct = 0; ct < 4; ++ct) {
        const int c = c0 + ct * 16 + l;
        const __bf16* vp = vb + (size_t)c * NN + m0 + sub * 32 + (hi ? 8 : 0);
        if (m0 + 128 < NN) __builtin_prefetch(vp + 128, 0, 1);
        const v16bf vB = ld_pair(vp, 16);
        oacc[0][ct] = __builtin_amdgcn_wmma_f32_16x16x32_bf16(false, pa0, false, vB,
                                                              (short)0, oacc[0][ct], false, false);
        oacc[1][ct] = __builtin_amdgcn_wmma_f32_16x16x32_bf16(false, pa1, false, vB,
                                                              (short)0, oacc[1][ct], false, false);
      }
    }
  }

  // ---- Epilogue: out = gamma * softmax_out + x ----
#pragma unroll
  for (int t = 0; t < 2; ++t) {
    const int nr = n0 + t * 16 + (hi ? 8 : 0);
#pragma unroll
    for (int ct = 0; ct < 4; ++ct) {
      const int c = c0 + ct * 16 + l;
      const float* xs = x   + ((size_t)b * CC + c) * NN + nr;
      float*       os = out + ((size_t)b * CC + c) * NN + nr;
#pragma unroll
      for (int e = 0; e < 8; ++e)
        os[e] = g * (oacc[t][ct][e] / rsum[t][e]) + xs[e];
    }
  }
}

extern "C" void kernel_launch(void* const* d_in, const int* in_sizes, int n_in,
                              void* d_out, int out_size, void* d_ws, size_t ws_size,
                              hipStream_t stream) {
  const float* x     = (const float*)d_in[0];
  const float* Wq    = (const float*)d_in[1];
  const float* bq    = (const float*)d_in[2];
  const float* Wk    = (const float*)d_in[3];
  const float* bk    = (const float*)d_in[4];
  const float* Wv    = (const float*)d_in[5];
  const float* bv    = (const float*)d_in[6];
  const float* gamma = (const float*)d_in[7];
  float* out = (float*)d_out;

  // Workspace: bf16 Q [B][N][32], K^T [B][N][32], V [B][C][N] -> 10.5 MB
  __bf16* Qw = (__bf16*)d_ws;
  __bf16* Kw = Qw + (size_t)BB * NN * CKK;
  __bf16* Vw = Kw + (size_t)BB * NN * CKK;

  proj_kernel<<<dim3(64, BB), 256, 0, stream>>>(x, Wq, bq, Qw, CKK, 1);
  proj_kernel<<<dim3(64, BB), 256, 0, stream>>>(x, Wk, bk, Kw, CKK, 1);
  proj_kernel<<<dim3(512, BB), 256, 0, stream>>>(x, Wv, bv, Vw, CC, 0);

  attn_kernel<<<dim3(NN / 32, BB), 128, 0, stream>>>(x, Qw, Kw, Vw, gamma, out);
}
